// SlotAttention_46110768890284
// MI455X (gfx1250) — compile-verified, matrix operand order checked by reading, not run on hardware
//
#include <hip/hip_runtime.h>
#include <hip/hip_bf16.h>

// ---------------- problem constants ----------------
constexpr int  kB   = 32;
constexpr int  kN   = 4096;
constexpr int  kS   = 16;
constexpr int  kH   = 4;
constexpr int  kIT  = 3;
constexpr float kSCALE  = 0.125f;   // DH^-0.5
constexpr float kEPS    = 1e-8f;
constexpr float kLNEPS  = 1e-5f;

// d_out layout (fp32): [slots 32*16*256][attn_last 32*16*4096][stack 3*32*16*4096]
constexpr long OUT_ATTN  = 131072L;
constexpr long OUT_STACK = 131072L + 2097152L;

typedef __attribute__((ext_vector_type(16))) __bf16 v16bf;
typedef __attribute__((ext_vector_type(8)))  float  v8f;
typedef __attribute__((ext_vector_type(4)))  int    v4i;

#if __has_builtin(__builtin_amdgcn_global_load_async_to_lds_b128) && \
    __has_builtin(__builtin_amdgcn_s_wait_asynccnt)
#define USE_ASYNC_LDS 1
#else
#define USE_ASYNC_LDS 0
#endif

static __device__ inline __bf16 f2bf(float f) {
  union { float f; unsigned u; } a; a.f = f;
  unsigned r = (a.u + 0x7FFFu + ((a.u >> 16) & 1u)) >> 16;   // RNE
  union { unsigned short s; __bf16 b; } o; o.s = (unsigned short)r;
  return o.b;
}

static __device__ inline v8f zero8() {
  v8f z;
  #pragma unroll
  for (int i = 0; i < 8; ++i) z[i] = 0.f;
  return z;
}

static __device__ inline v8f wmma_bf16(v16bf a, v16bf b, v8f c) {
  return __builtin_amdgcn_wmma_f32_16x16x32_bf16(false, a, false, b, (short)0, c,
                                                 false, false);
}

// A fragment 16x32 (MxK), row-major with leading dim ld (bf16 elements).
// CDNA5 layout: lanes 0-15 hold M=0..15 / K=0-7,16-23, lanes 16-31 K=8-15,24-31.
static __device__ inline v16bf ld_frag_mk(const __bf16* p, int ld) {
  int lane = threadIdx.x & 31;
  int m = lane & 15, hv = lane >> 4;
  const __bf16* r = p + m * ld + hv * 8;
  v16bf f;
  #pragma unroll
  for (int j = 0; j < 8; ++j) {
    int k = ((j >> 2) << 4) + ((j & 3) << 1);
    f[2 * j]     = r[k];
    f[2 * j + 1] = r[k + 1];
  }
  return f;
}

// B fragment 32x16 (KxN); element (k,n) at p[k*ldk + n*ldn]. Lane holds column n.
static __device__ inline v16bf ld_frag_kn(const __bf16* p, long ldk, long ldn) {
  int lane = threadIdx.x & 31;
  int n = lane & 15, hv = lane >> 4;
  const __bf16* c = p + (long)n * ldn + (long)(hv * 8) * ldk;
  v16bf f;
  #pragma unroll
  for (int j = 0; j < 8; ++j) {
    long k = (long)(((j >> 2) << 4) + ((j & 3) << 1));
    f[2 * j]     = c[k * ldk];
    f[2 * j + 1] = c[(k + 1) * ldk];
  }
  return f;
}

// ---------------- prep kernels ----------------
__global__ void convert_weights_kernel(const float* Wq, const float* Wk, const float* Wv,
                                       __bf16* wq, __bf16* wk, __bf16* wv) {
  int i = blockIdx.x * 256 + threadIdx.x;        // 3*65536 total
  int which = i >> 16, idx = i & 65535;
  if (which == 0)      wq[idx] = f2bf(Wq[idx]);
  else if (which == 1) wk[idx] = f2bf(Wk[idx]);
  else                 wv[idx] = f2bf(Wv[idx]);
}

__global__ void init_slots_kernel(const float* cond, float* slots) {
  int i = blockIdx.x * 256 + threadIdx.x;        // 131072
  slots[i] = cond[i];
}

__global__ void zero_rowsum_kernel(float* rowsum) {
  int i = blockIdx.x * 256 + threadIdx.x;        // 2048
  rowsum[i] = 0.f;
}

// ---------------- LN(inputs) + K/V projection (bulk GEMM) ----------------
// k stored natural [b*N+n][c] (bf16); v stored transposed v_t[((b*4+h)*64+dh)][n].
__global__ void __launch_bounds__(256)
ln_kv_kernel(const float* __restrict__ x, const float* __restrict__ ln_g,
             const float* __restrict__ ln_b, const __bf16* __restrict__ wkw,
             const __bf16* __restrict__ wvw, __bf16* __restrict__ k_bf,
             __bf16* __restrict__ v_t) {
  __shared__ __bf16 xs[64][264];                 // 64 rows of LN'd input, bf16
  int tid = threadIdx.x, lane = tid & 31, wave = tid >> 5;
  long row0 = (long)blockIdx.x * 64;

  // layer-norm: each wave owns 8 rows; 32-lane reductions (wave32)
  for (int i = 0; i < 8; ++i) {
    int r = wave * 8 + i;
    const float* xp = x + (row0 + r) * 256;
    float vreg[8], s = 0.f, s2 = 0.f;
    #pragma unroll
    for (int j = 0; j < 8; ++j) {
      float t = xp[lane + 32 * j];
      vreg[j] = t; s += t; s2 += t * t;
    }
    #pragma unroll
    for (int o = 16; o > 0; o >>= 1) {
      s  += __shfl_xor(s,  o, 32);
      s2 += __shfl_xor(s2, o, 32);
    }
    float mu = s * (1.f / 256.f);
    float var = s2 * (1.f / 256.f) - mu * mu;
    float rstd = rsqrtf(var + kLNEPS);
    #pragma unroll
    for (int j = 0; j < 8; ++j) {
      int c = lane + 32 * j;
      xs[r][c] = f2bf((vreg[j] - mu) * rstd * ln_g[c] + ln_b[c]);
    }
  }
  __syncthreads();

  // GEMM: 64 rows x 512 cols (k:0..255, v:256..511), K=256
  int cg = wave * 64;                            // this wave's 64 output cols
  v8f acc[4][4];
  #pragma unroll
  for (int mt = 0; mt < 4; ++mt)
    #pragma unroll
    for (int nt = 0; nt < 4; ++nt) acc[mt][nt] = zero8();

  #pragma unroll
  for (int kk = 0; kk < 8; ++kk) {
    v16bf bfr[4];
    #pragma unroll
    for (int nt = 0; nt < 4; ++nt) {
      int c0 = cg + nt * 16;
      const __bf16* wb = (c0 < 256) ? (wkw + c0 * 256) : (wvw + (c0 - 256) * 256);
      bfr[nt] = ld_frag_kn(wb + kk * 32, 1, 256);    // B[k=f][n=c] = W[c][f]
    }
    #pragma unroll
    for (int mt = 0; mt < 4; ++mt) {
      v16bf af = ld_frag_mk(&xs[mt * 16][kk * 32], 264);
      #pragma unroll
      for (int nt = 0; nt < 4; ++nt)
        acc[mt][nt] = wmma_bf16(af, bfr[nt], acc[mt][nt]);
    }
  }

  int nl = lane & 15, hv = lane >> 4;
  #pragma unroll
  for (int mt = 0; mt < 4; ++mt)
    #pragma unroll
    for (int nt = 0; nt < 4; ++nt) {
      int c0 = cg + nt * 16;
      if (c0 < 256) {                    // k: natural layout
        int cc = c0 + nl;
        #pragma unroll
        for (int r = 0; r < 8; ++r) {
          long grow = row0 + mt * 16 + r + 8 * hv;
          k_bf[grow * 256 + cc] = f2bf(acc[mt][nt][r]);
        }
      } else {                           // v: transposed [ (b*4+h)*64+dh ][ n ]
        int cc = (c0 - 256) + nl;
        int h = cc >> 6, d = cc & 63;
        #pragma unroll
        for (int r = 0; r < 8; ++r) {
          long grow = row0 + mt * 16 + r + 8 * hv;
          long bb = grow >> 12, n = grow & 4095;
          v_t[(((bb * 4) + h) * 64 + d) * 4096 + n] = f2bf(acc[mt][nt][r]);
        }
      }
    }
}

// ---------------- slots LN + q = LN(slots) @ Wq^T * scale ----------------
__global__ void __launch_bounds__(128)
qproj_kernel(const float* __restrict__ slots, const float* __restrict__ ln_g,
             const float* __restrict__ ln_b, const __bf16* __restrict__ wqw,
             __bf16* __restrict__ q_bf) {
  __shared__ __bf16 ss[16][264];
  int b = blockIdx.x, tid = threadIdx.x, lane = tid & 31, wave = tid >> 5;

  for (int i = 0; i < 4; ++i) {
    int r = wave * 4 + i;
    const float* sp = slots + ((long)b * 16 + r) * 256;
    float vreg[8], s = 0.f, s2 = 0.f;
    #pragma unroll
    for (int j = 0; j < 8; ++j) {
      float t = sp[lane + 32 * j];
      vreg[j] = t; s += t; s2 += t * t;
    }
    #pragma unroll
    for (int o = 16; o > 0; o >>= 1) {
      s  += __shfl_xor(s,  o, 32);
      s2 += __shfl_xor(s2, o, 32);
    }
    float mu = s * (1.f / 256.f);
    float var = s2 * (1.f / 256.f) - mu * mu;
    float rstd = rsqrtf(var + kLNEPS);
    #pragma unroll
    for (int j = 0; j < 8; ++j) {
      int c = lane + 32 * j;
      ss[r][c] = f2bf((vreg[j] - mu) * rstd * ln_g[c] + ln_b[c]);
    }
  }
  __syncthreads();

  v8f acc[4];
  #pragma unroll
  for (int nt = 0; nt < 4; ++nt) acc[nt] = zero8();
  #pragma unroll
  for (int kk = 0; kk < 8; ++kk) {
    v16bf af = ld_frag_mk(&ss[0][kk * 32], 264);
    #pragma unroll
    for (int nt = 0; nt < 4; ++nt) {
      int c0 = wave * 64 + nt * 16;
      acc[nt] = wmma_bf16(af, ld_frag_kn(wqw + c0 * 256 + kk * 32, 1, 256), acc[nt]);
    }
  }
  int nl = lane & 15, hv = lane >> 4;
  #pragma unroll
  for (int nt = 0; nt < 4; ++nt) {
    int c0 = wave * 64 + nt * 16;
    #pragma unroll
    for (int r = 0; r < 8; ++r) {
      int srow = r + 8 * hv;
      int c = c0 + nl, h = c >> 6, d = c & 63;
      q_bf[(((long)b * 4 + h) * 16 + srow) * 64 + d] = f2bf(acc[nt][r] * kSCALE);
    }
  }
}

// ------- dots (WMMA) + slot-competition softmax + outputs + rowsum -------
__global__ void __launch_bounds__(256)
attn_kernel(const __bf16* __restrict__ q_bf, const __bf16* __restrict__ k_bf,
            __bf16* __restrict__ attn_bf, float* __restrict__ rowsum,
            float* __restrict__ out, int it) {
  __shared__ float  ds[64][132];          // [s*4+h][128 cols], padded
  __shared__ __bf16 qs[4][16][80];        // q per head; stride 160B (16B-aligned)
  int tid = threadIdx.x, lane = tid & 31, wave = tid >> 5;
  int b = blockIdx.x >> 5;
  int n0 = (blockIdx.x & 31) * 128;

#if USE_ASYNC_LDS
  {
    int i = tid;                          // 512 16-byte chunks, 2 per thread
    #pragma unroll
    for (int rep = 0; rep < 2; ++rep, i += 256) {
      int hs = i >> 3, j = i & 7;         // hs = h*16+s in [0,64), j = 8-elem chunk
      const __bf16* src = q_bf + ((long)b * 64 + hs) * 64 + j * 8;
      __bf16* dst = &qs[0][0][0] + hs * 80 + j * 8;
      __builtin_amdgcn_global_load_async_to_lds_b128(
          (__attribute__((address_space(1))) v4i*)const_cast<__bf16*>(src),
          (__attribute__((address_space(3))) v4i*)dst, 0, 0);
    }
  }
  __builtin_amdgcn_s_wait_asynccnt(0);
  __syncthreads();
#else
  for (int i = tid; i < 4096; i += 256) {
    int h = i >> 10, rem = i & 1023, s = rem >> 6, d = rem & 63;
    qs[h][s][d] = q_bf[(((long)b * 4 + h) * 16 + s) * 64 + d];
  }
  __syncthreads();
#endif

  int h = wave >> 1, nh = (wave & 1) * 64;     // wave -> (head, 64-col half)
  v8f acc[4];
  #pragma unroll
  for (int nt = 0; nt < 4; ++nt) acc[nt] = zero8();
  #pragma unroll
  for (int kk = 0; kk < 2; ++kk) {             // K = DH = 64
    v16bf af = ld_frag_mk(&qs[h][0][kk * 32], 80);
    #pragma unroll
    for (int nt = 0; nt < 4; ++nt) {
      long col0 = (long)n0 + nh + nt * 16;
      const __bf16* kb = k_bf + ((long)b * 4096 + col0) * 256 + h * 64 + kk * 32;
      acc[nt] = wmma_bf16(af, ld_frag_kn(kb, 1, 256), acc[nt]);
    }
  }
  int nl = lane & 15, hv = lane >> 4;
  #pragma unroll
  for (int nt = 0; nt < 4; ++nt)
    #pragma unroll
    for (int r = 0; r < 8; ++r) {
      int s = r + 8 * hv;
      ds[s * 4 + h][nh + nt * 16 + nl] = acc[nt][r];
    }
  __syncthreads();

  // softmax over the 64 (s*h) rows, per column n — fully local to this tile
  if (tid < 128) {
    int j = tid;
    float mx = -3.4e38f;
    for (int r = 0; r < 64; ++r) mx = fmaxf(mx, ds[r][j]);
    float sum = 0.f;
    for (int r = 0; r < 64; ++r) { float e = __expf(ds[r][j] - mx); ds[r][j] = e; sum += e; }
    float inv = 1.f / sum;
    for (int r = 0; r < 64; ++r) ds[r][j] *= inv;          // attn_before
    for (int s = 0; s < 16; ++s) {                         // head-mean output
      float m4 = (ds[s*4][j] + ds[s*4+1][j] + ds[s*4+2][j] + ds[s*4+3][j]) * 0.25f;
      long base = ((long)b * 16 + s) * 4096 + n0 + j;
      out[OUT_STACK + (long)it * 2097152L + base] = m4;
      if (it == kIT - 1) out[OUT_ATTN + base] = m4;
    }
  }
  __syncthreads();

  for (int i = tid; i < 8192; i += 256) {
    int r = i >> 7, j = i & 127, s = r >> 2, hh = r & 3;
    attn_bf[(((long)b * 4 + hh) * 16 + s) * 4096 + n0 + j] = f2bf(ds[r][j]);
  }
  { // per-(s,h) row sums -> global accumulation for second normalization
    int r = tid >> 2, qt = tid & 3;
    float sp = 0.f;
    for (int j = 0; j < 32; ++j) sp += ds[r][qt * 32 + j];
    sp += __shfl_xor(sp, 1, 32);
    sp += __shfl_xor(sp, 2, 32);
    if (qt == 0) atomicAdd(&rowsum[b * 64 + r], sp);
  }
}

// ---------------- updates = (attn/rowsum) @ v (WMMA over K=N) ----------------
__global__ void __launch_bounds__(256)
updates_kernel(const __bf16* __restrict__ attn_bf, const __bf16* __restrict__ v_t,
               const float* __restrict__ rowsum, float* __restrict__ updates) {
  __shared__ float ps[8][16][64];
  int tid = threadIdx.x, lane = tid & 31, wave = tid >> 5;
  int b = blockIdx.x >> 2, h = blockIdx.x & 3;

  v8f acc[4];
  #pragma unroll
  for (int nt = 0; nt < 4; ++nt) acc[nt] = zero8();
  const __bf16* abase = attn_bf + (((long)b * 4 + h) * 16) * 4096;
  const __bf16* vtb   = v_t + (((long)b * 4 + h) * 64) * 4096;
  for (int ks = 0; ks < 16; ++ks) {            // each wave covers 512 of K=4096
    long kbase = (long)wave * 512 + ks * 32;
    if (ks + 1 < 16) {                         // prefetch next streaming chunk
      __builtin_prefetch(abase + kbase + 32 + (lane & 15) * 4096, 0, 1);
      __builtin_prefetch(vtb + (long)(lane & 15) * 4096 + kbase + 32, 0, 1);
    }
    v16bf af = ld_frag_mk(abase + kbase, 4096);
    #pragma unroll
    for (int nt = 0; nt < 4; ++nt) {
      const __bf16* vb = vtb + (long)(nt * 16) * 4096 + kbase;   // contiguous per lane
      acc[nt] = wmma_bf16(af, ld_frag_kn(vb, 1, 4096), acc[nt]);
    }
  }
  int nl = lane & 15, hv = lane >> 4;
  #pragma unroll
  for (int nt = 0; nt < 4; ++nt)
    #pragma unroll
    for (int r = 0; r < 8; ++r)
      ps[wave][r + 8 * hv][nt * 16 + nl] = acc[nt][r];
  __syncthreads();

  for (int i = tid; i < 1024; i += 256) {
    int s = i >> 6, d = i & 63;
    float sum = 0.f;
    #pragma unroll
    for (int w = 0; w < 8; ++w) sum += ps[w][s][d];
    float rinv = 1.f / (rowsum[b * 64 + s * 4 + h] + kEPS);
    updates[((long)b * 16 + s) * 256 + h * 64 + d] = sum * rinv;
  }
}

// ---------------- GRU cell (tiny; fp32 VALU) ----------------
__global__ void __launch_bounds__(256)
gru_kernel(const float* __restrict__ updates, float* __restrict__ slots,
           const float* __restrict__ W_ih, const float* __restrict__ W_hh,
           const float* __restrict__ b_ih, const float* __restrict__ b_hh,
           float* __restrict__ out, int it) {
  __shared__ float xr[256], hr[256];
  int row = blockIdx.x, tid = threadIdx.x;
  xr[tid] = updates[(long)row * 256 + tid];
  hr[tid] = slots[(long)row * 256 + tid];
  __syncthreads();
  float gi[3], gh[3];
  #pragma unroll
  for (int g = 0; g < 3; ++g) {
    int c = g * 256 + tid;
    float a = b_ih[c], bb = b_hh[c];
    const float* wi = W_ih + (long)c * 256;
    const float* wh = W_hh + (long)c * 256;
    for (int f = 0; f < 256; ++f) { a += xr[f] * wi[f]; bb += hr[f] * wh[f]; }
    gi[g] = a; gh[g] = bb;
  }
  float r = 1.f / (1.f + __expf(-(gi[0] + gh[0])));
  float z = 1.f / (1.f + __expf(-(gi[1] + gh[1])));
  float n = tanhf(gi[2] + r * gh[2]);
  float hnew = (1.f - z) * n + z * hr[tid];
  slots[(long)row * 256 + tid] = hnew;
  if (it == kIT - 1) out[(long)row * 256 + tid] = hnew;
}

// ---------------- launch ----------------
extern "C" void kernel_launch(void* const* d_in, const int* in_sizes, int n_in,
                              void* d_out, int out_size, void* d_ws, size_t ws_size,
                              hipStream_t stream) {
  (void)in_sizes; (void)n_in; (void)out_size; (void)ws_size;
  const float* inputs  = (const float*)d_in[0];
  const float* cond    = (const float*)d_in[1];
  const float* Wq      = (const float*)d_in[2];
  const float* Wk      = (const float*)d_in[3];
  const float* Wv      = (const float*)d_in[4];
  const float* ln_in_g = (const float*)d_in[5];
  const float* ln_in_b = (const float*)d_in[6];
  const float* ln_s_g  = (const float*)d_in[7];
  const float* ln_s_b  = (const float*)d_in[8];
  const float* W_ih    = (const float*)d_in[9];
  const float* W_hh    = (const float*)d_in[10];
  const float* b_ih    = (const float*)d_in[11];
  const float* b_hh    = (const float*)d_in[12];
  float* out = (float*)d_out;

  char* p = (char*)d_ws;
  __bf16* k_bf    = (__bf16*)p; p += (long)kB * kN * 256 * 2;     // 64 MB
  __bf16* v_t     = (__bf16*)p; p += (long)kB * kN * 256 * 2;     // 64 MB (transposed)
  __bf16* attn_bf = (__bf16*)p; p += (long)kB * 64 * kN * 2;      // 16 MB
  __bf16* q_bf    = (__bf16*)p; p += 131072L * 2;
  __bf16* wq_bf   = (__bf16*)p; p += 65536L * 2;
  __bf16* wk_bf   = (__bf16*)p; p += 65536L * 2;
  __bf16* wv_bf   = (__bf16*)p; p += 65536L * 2;
  float*  slots   = (float*)p;  p += 131072L * 4;
  float*  updates = (float*)p;  p += 131072L * 4;
  float*  rowsum  = (float*)p;  p += 2048L * 4;

  convert_weights_kernel<<<768, 256, 0, stream>>>(Wq, Wk, Wv, wq_bf, wk_bf, wv_bf);
  init_slots_kernel<<<512, 256, 0, stream>>>(cond, slots);
  ln_kv_kernel<<<(kB * kN) / 64, 256, 0, stream>>>(inputs, ln_in_g, ln_in_b,
                                                   wk_bf, wv_bf, k_bf, v_t);
  for (int it = 0; it < kIT; ++it) {
    zero_rowsum_kernel<<<8, 256, 0, stream>>>(rowsum);
    qproj_kernel<<<kB, 128, 0, stream>>>(slots, ln_s_g, ln_s_b, wq_bf, q_bf);
    attn_kernel<<<kB * (kN / 128), 256, 0, stream>>>(q_bf, k_bf, attn_bf, rowsum,
                                                     out, it);
    updates_kernel<<<kB * kH, 256, 0, stream>>>(attn_bf, v_t, rowsum, updates);
    gru_kernel<<<kB * kS, 256, 0, stream>>>(updates, slots, W_ih, W_hh, b_ih, b_hh,
                                            out, it);
  }
}
